// LanguageModel_79577154060918
// MI455X (gfx1250) — matrix operand order repaired and checked by reading, not compiled, not run on hardware
//
#include <hip/hip_runtime.h>

typedef _Float16 half_t;
typedef __attribute__((ext_vector_type(16))) _Float16 v16h;
typedef __attribute__((ext_vector_type(8)))  _Float16 v8h;
typedef __attribute__((ext_vector_type(8)))  float    v8f;

#define DEVFN static __device__ __forceinline__

// ---------------------------------------------------------------------------
// WMMA helpers: D = A(16x32 f16) * B(32x16 f16) + C(16x16 f32), wave32.
// A and B both use contiguous k order: k = 16*(lane/16) + element.
// ---------------------------------------------------------------------------
DEVFN v8f wmma16(v16h a, v16h b, v8f c) {
  return __builtin_amdgcn_wmma_f32_16x16x32_f16(
      /*neg_a=*/false, a, /*neg_b=*/false, b,
      /*c_mod=*/(short)0, c, /*reuse_a=*/false, /*reuse_b=*/false);
}

// A fragment: base points at element [row0, k0] of a row-major f16 matrix with
// leading dimension ld (halves).  Lane L holds row (L&15); kb = (L<16)?0:16.
// VGPR0-3 <- k = kb..kb+7 ; VGPR4-7 <- k = kb+8..kb+15.
DEVFN v16h load_A(const half_t* base, int ld) {
  const int lane = threadIdx.x & 31;
  const half_t* p = base + (lane & 15) * ld + ((lane >> 4) << 4);
  v8h lo = *(const v8h*)p;          // 16B
  v8h hi = *(const v8h*)(p + 8);    // 16B
  v16h a;
#pragma unroll
  for (int i = 0; i < 8; ++i) { a[i] = lo[i]; a[8 + i] = hi[i]; }
  return a;
}

// B fragment: B = W^T where W is [N][K] row-major f16 with ld = K-stride.
// w points at row n0 of W.  Lane L holds col n = L&15 and 16 contiguous k's
// starting at k0 + ((L<16)?0:16)  -> single 32B load.
DEVFN v16h load_B(const half_t* w, int ld, int k0) {
  const int lane = threadIdx.x & 31;
  const half_t* p = w + (lane & 15) * ld + k0 + ((lane >> 4) << 4);
  return *(const v16h*)p;           // 32B
}

// C fragment -> row-major f32 [16][ldc]
DEVFN void store_C(float* dst, int ldc, v8f c) {
  const int lane = threadIdx.x & 31;
  const int n = lane & 15;
  const int mb = (lane < 16) ? 0 : 8;
#pragma unroll
  for (int v = 0; v < 8; ++v) dst[(mb + v) * ldc + n] = c[v];
}

// C init = broadcast bias over rows; b points at bias[col0]
DEVFN v8f bias_C(const float* b) {
  const float bv = b[threadIdx.x & 15];
  v8f c;
#pragma unroll
  for (int v = 0; v < 8; ++v) c[v] = bv;
  return c;
}

DEVFN float sigm(float x)  { return 1.0f / (1.0f + __expf(-x)); }
DEVFN float tanh_f(float x){ float e = __expf(2.0f * x); return (e - 1.0f) / (e + 1.0f); }
DEVFN float relu_f(float x){ return x > 0.0f ? x : 0.0f; }

// ---------------------------------------------------------------------------
// f32 -> f16 weight conversion
// ---------------------------------------------------------------------------
__global__ void f32_to_f16_kernel(const float* __restrict__ src,
                                  half_t* __restrict__ dst, int n) {
  int i = blockIdx.x * blockDim.x + threadIdx.x;
  if (i < n) dst[i] = (half_t)src[i];
}

// ---------------------------------------------------------------------------
// Char-level word embedder: 16 words / block, 8 waves, LDS 157.8KB
// (2 workgroups / WGP).  2-layer biLSTM (H=64/dir, in=128) + MLP 128->128->128.
// Weight B-fragments + biases live in registers across the 16-step scan.
// ---------------------------------------------------------------------------
__global__ void __launch_bounds__(256)
word_lstm_kernel(const int* __restrict__ sent_tok,    // [8192][16]
                 const int* __restrict__ word_tok,    // [128][16]
                 const float* __restrict__ emb,       // [129][128] f32
                 const half_t* __restrict__ wih_g,    // [2][2][256][128] f16
                 const half_t* __restrict__ whh_g,    // [2][2][256][64]  f16
                 const float*  __restrict__ bias_g,   // [2][2][256]      f32
                 const half_t* __restrict__ fcW1, const float* __restrict__ fcb1,
                 const half_t* __restrict__ fcW2, const float* __restrict__ fcb2,
                 half_t* __restrict__ word_emb)       // [8320][128] f16
{
  extern __shared__ char smem[];
  half_t* x0   = (half_t*)smem;            // [16 w][16 t][128]  32768 h
  half_t* seqf = x0   + 32768;             // [16][16][64]       16384 h
  half_t* seqb = seqf + 16384;             // 16384 h
  half_t* hbuf = seqb + 16384;             // [16][64]           1024 h
  half_t* hfin = hbuf + 1024;              // [16][128]          2048 h
  float*  gates= (float*)(hfin + 2048);    // [16][256]          4096 f
  float*  cst  = gates + 4096;             // [16][64]           1024 f
  int*    ne   = (int*)(cst + 1024);       // [16]

  const int tid = threadIdx.x, wave = tid >> 5, lane = tid & 31;
  const int word0 = blockIdx.x * 16;

  // Stage char embeddings -> x0 (one row = one (word, t) pair per thread)
  {
    const int m = tid >> 4, t = tid & 15;
    const int w = word0 + m;
    const int* tok = (w < 8192) ? (sent_tok + (size_t)w * 16)
                                : (word_tok + (size_t)(w - 8192) * 16);
    const int c = tok[t];
    const float4* e4 = (const float4*)(emb + (size_t)c * 128);
    half_t* dst = x0 + (size_t)tid * 128;
#pragma unroll 8
    for (int d = 0; d < 32; ++d) {
      const float4 v = e4[d];
      dst[d * 4 + 0] = (half_t)v.x; dst[d * 4 + 1] = (half_t)v.y;
      dst[d * 4 + 2] = (half_t)v.z; dst[d * 4 + 3] = (half_t)v.w;
    }
  }
  if (tid < 16) {
    const int w = word0 + tid;
    const int* tok = (w < 8192) ? (sent_tok + (size_t)w * 16)
                                : (word_tok + (size_t)(w - 8192) * 16);
    int any = 0;
#pragma unroll
    for (int t = 0; t < 16; ++t) any |= tok[t];
    ne[tid] = (any != 0) ? 1 : 0;
  }
  __syncthreads();

  for (int layer = 0; layer < 2; ++layer) {
    for (int dir = 0; dir < 2; ++dir) {
      const int ld_idx = layer * 2 + dir;
      const half_t* wsrc = wih_g  + (size_t)ld_idx * 256 * 128;
      const half_t* hsrc = whh_g  + (size_t)ld_idx * 256 * 64;
      const float*  bsrc = bias_g + (size_t)ld_idx * 256;

      // Prefetch next (layer,dir) weights into L2 (global_prefetch_b8)
      {
        const int nxt = (ld_idx + 1) & 3;
        const char* pw = (const char*)(wih_g + (size_t)nxt * 256 * 128);
        const char* ph = (const char*)(whh_g + (size_t)nxt * 256 * 64);
        for (int pf = lane; pf < 64; pf += 32) __builtin_prefetch(pw + pf * 1024, 0, 1);
        for (int pf = lane; pf < 32; pf += 32) __builtin_prefetch(ph + pf * 1024, 0, 1);
      }

      // Hoist weight fragments + bias into registers (loop-invariant over t)
      v16h Bi[2][4], Bh[2][2];
      v8f  bias[2];
#pragma unroll
      for (int q = 0; q < 2; ++q) {
        const int cb = (wave * 2 + q) * 16;
#pragma unroll
        for (int kc = 0; kc < 4; ++kc)
          Bi[q][kc] = load_B(wsrc + (size_t)cb * 128, 128, kc * 32);
#pragma unroll
        for (int kc = 0; kc < 2; ++kc)
          Bh[q][kc] = load_B(hsrc + (size_t)cb * 64, 64, kc * 32);
        bias[q] = bias_C(bsrc + cb);
      }

      for (int i = tid; i < 1024; i += 256) { cst[i] = 0.0f; hbuf[i] = (half_t)0.0f; }
      __syncthreads();

      half_t* seq = dir ? seqb : seqf;
      for (int step = 0; step < 16; ++step) {
        const int t = dir ? (15 - step) : step;
        // gates[16][256] = x_t @ Wih^T + h @ Whh^T + b ; 2 col-tiles per wave
#pragma unroll
        for (int q = 0; q < 2; ++q) {
          v8f c = bias[q];
#pragma unroll
          for (int kc = 0; kc < 4; ++kc)              // input K=128
            c = wmma16(load_A(x0 + t * 128 + kc * 32, 16 * 128), Bi[q][kc], c);
#pragma unroll
          for (int kc = 0; kc < 2; ++kc)              // hidden K=64
            c = wmma16(load_A(hbuf + kc * 32, 64), Bh[q][kc], c);
          store_C(gates + (wave * 2 + q) * 16, 256, c);
        }
        __syncthreads();
        // pointwise LSTM cell: 16 x 64 elements
#pragma unroll
        for (int p = 0; p < 4; ++p) {
          const int e = tid + p * 256;
          const int m = e >> 6, j = e & 63;
          const float gi = gates[m * 256 + j];
          const float gf = gates[m * 256 + 64 + j];
          const float gg = gates[m * 256 + 128 + j];
          const float go = gates[m * 256 + 192 + j];
          const float cm = sigm(gf) * cst[e] + sigm(gi) * tanh_f(gg);
          cst[e] = cm;
          const float h = sigm(go) * tanh_f(cm);
          hbuf[e] = (half_t)h;
          seq[(m * 16 + t) * 64 + j] = (half_t)h;
        }
        __syncthreads();
      }
      // record final hidden (layer 1 overwrites layer 0 -> last layer wins)
      for (int e = tid; e < 1024; e += 256)
        hfin[(e >> 6) * 128 + dir * 64 + (e & 63)] = hbuf[e];
      __syncthreads();
    }
    if (layer == 0) {
      // next-layer input = concat(seqf, seqb); one row per thread
      half_t* dst = x0 + (size_t)tid * 128;
      const half_t* sf = seqf + (size_t)tid * 64;
      const half_t* sb = seqb + (size_t)tid * 64;
#pragma unroll 4
      for (int j = 0; j < 64; ++j) { dst[j] = sf[j]; dst[64 + j] = sb[j]; }
      __syncthreads();
    }
  }

  // MLP: h1 = relu(hfin @ W1^T + b1); we = h1 @ W2^T + b2; mask; store f16
  half_t* h1 = x0;  // reuse
  {
    const int cb = wave * 16;
    v8f c = bias_C(fcb1 + cb);
#pragma unroll
    for (int kc = 0; kc < 4; ++kc)
      c = wmma16(load_A(hfin + kc * 32, 128),
                 load_B(fcW1 + (size_t)cb * 128, 128, kc * 32), c);
    const int n = lane & 15, mb = (lane < 16) ? 0 : 8;
#pragma unroll
    for (int v = 0; v < 8; ++v)
      h1[(mb + v) * 128 + cb + n] = (half_t)relu_f(c[v]);
  }
  __syncthreads();
  {
    const int cb = wave * 16;
    v8f c = bias_C(fcb2 + cb);
#pragma unroll
    for (int kc = 0; kc < 4; ++kc)
      c = wmma16(load_A(h1 + kc * 32, 128),
                 load_B(fcW2 + (size_t)cb * 128, 128, kc * 32), c);
    const int n = lane & 15, mb = (lane < 16) ? 0 : 8;
#pragma unroll
    for (int v = 0; v < 8; ++v) {
      const int m = mb + v;
      const float x = ne[m] ? c[v] : 0.0f;
      word_emb[(size_t)(word0 + m) * 128 + cb + n] = (half_t)x;
    }
  }
}

// ---------------------------------------------------------------------------
// Sentence-level biLSTM: 16 sentences / block, 16 waves, LDS 61.4KB.
// 2-layer biLSTM (H=128/dir; in 128 then 256) + MLP 256->256->256.
// All weight fragments register-resident across the 64-step scan.
// ---------------------------------------------------------------------------
__global__ void __launch_bounds__(512)
sent_lstm_kernel(const half_t* __restrict__ word_emb,  // [8192..][128] f16
                 const half_t* __restrict__ wih0,      // [2][512][128]
                 const half_t* __restrict__ wih1,      // [2][512][256]
                 const half_t* __restrict__ whh_g,     // [2][2][512][128]
                 const float*  __restrict__ bias_g,    // [2][2][512]
                 const half_t* __restrict__ fcW1, const float* __restrict__ fcb1,
                 const half_t* __restrict__ fcW2, const float* __restrict__ fcb2,
                 half_t* __restrict__ seq_ws,          // [128][64][256] f16
                 half_t* __restrict__ sent_emb)        // [128][256] f16
{
  extern __shared__ char smem[];
  half_t* hbuf = (half_t*)smem;            // [16][128]   2048 h
  half_t* hfin = hbuf + 2048;              // [16][256]   4096 h
  half_t* h1   = hfin + 4096;              // [16][256]   4096 h
  float*  gates= (float*)(h1 + 4096);      // [16][512]   8192 f
  float*  cst  = gates + 8192;             // [16][128]   2048 f

  const int tid = threadIdx.x, wave = tid >> 5, lane = tid & 31;
  const int b0 = blockIdx.x * 16;

  for (int layer = 0; layer < 2; ++layer) {
    for (int dir = 0; dir < 2; ++dir) {
      const int ld_idx = layer * 2 + dir;
      const half_t* wih = layer ? (wih1 + (size_t)dir * 512 * 256)
                                : (wih0 + (size_t)dir * 512 * 128);
      const half_t* hsrc = whh_g + (size_t)ld_idx * 512 * 128;
      const float*  bsrc = bias_g + (size_t)ld_idx * 512;

      // Hoist weight fragments + bias into registers (loop-invariant over t)
      v16h Bi[2][8], Bh[2][4];
      v8f  bias[2];
#pragma unroll
      for (int q = 0; q < 2; ++q) {
        const int cb = (wave * 2 + q) * 16;   // 32 col tiles over 512 gates
        if (layer == 0) {
#pragma unroll
          for (int kc = 0; kc < 4; ++kc)
            Bi[q][kc] = load_B(wih + (size_t)cb * 128, 128, kc * 32);
        } else {
#pragma unroll
          for (int kc = 0; kc < 8; ++kc)
            Bi[q][kc] = load_B(wih + (size_t)cb * 256, 256, kc * 32);
        }
#pragma unroll
        for (int kc = 0; kc < 4; ++kc)
          Bh[q][kc] = load_B(hsrc + (size_t)cb * 128, 128, kc * 32);
        bias[q] = bias_C(bsrc + cb);
      }

      for (int i = tid; i < 2048; i += 512) { cst[i] = 0.0f; hbuf[i] = (half_t)0.0f; }
      __syncthreads();

      for (int step = 0; step < 64; ++step) {
        const int t = dir ? (63 - step) : step;
#pragma unroll
        for (int q = 0; q < 2; ++q) {
          v8f c = bias[q];
          if (layer == 0) {
            const half_t* xb = word_emb + ((size_t)b0 * 64 + t) * 128;
#pragma unroll
            for (int kc = 0; kc < 4; ++kc)
              c = wmma16(load_A(xb + kc * 32, 64 * 128), Bi[q][kc], c);
          } else {
            const half_t* xb = seq_ws + ((size_t)b0 * 64 + t) * 256;
#pragma unroll
            for (int kc = 0; kc < 8; ++kc)
              c = wmma16(load_A(xb + kc * 32, 64 * 256), Bi[q][kc], c);
          }
#pragma unroll
          for (int kc = 0; kc < 4; ++kc)        // hidden K=128
            c = wmma16(load_A(hbuf + kc * 32, 128), Bh[q][kc], c);
          store_C(gates + (wave * 2 + q) * 16, 512, c);
        }
        __syncthreads();
#pragma unroll
        for (int p = 0; p < 4; ++p) {           // 16 x 128 pointwise
          const int e = tid + p * 512;
          const int m = e >> 7, j = e & 127;
          const float gi = gates[m * 512 + j];
          const float gf = gates[m * 512 + 128 + j];
          const float gg = gates[m * 512 + 256 + j];
          const float go = gates[m * 512 + 384 + j];
          const float cm = sigm(gf) * cst[e] + sigm(gi) * tanh_f(gg);
          cst[e] = cm;
          const float h = sigm(go) * tanh_f(cm);
          hbuf[e] = (half_t)h;
          if (layer == 0)
            seq_ws[(((size_t)(b0 + m)) * 64 + t) * 256 + dir * 128 + j] = (half_t)h;
        }
        __syncthreads();
      }
      if (layer == 0 && dir == 1) { __threadfence(); __syncthreads(); }
      if (layer == 1) {
        for (int e = tid; e < 2048; e += 512)
          hfin[(e >> 7) * 256 + dir * 128 + (e & 127)] = hbuf[e];
        __syncthreads();
      }
    }
  }

  // MLP 256 -> relu -> 256 ; one col tile per wave
  {
    const int cb = wave * 16;
    v8f c = bias_C(fcb1 + cb);
#pragma unroll
    for (int kc = 0; kc < 8; ++kc)
      c = wmma16(load_A(hfin + kc * 32, 256),
                 load_B(fcW1 + (size_t)cb * 256, 256, kc * 32), c);
    const int n = lane & 15, mb = (lane < 16) ? 0 : 8;
#pragma unroll
    for (int v = 0; v < 8; ++v)
      h1[(mb + v) * 256 + cb + n] = (half_t)relu_f(c[v]);
  }
  __syncthreads();
  {
    const int cb = wave * 16;
    v8f c = bias_C(fcb2 + cb);
#pragma unroll
    for (int kc = 0; kc < 8; ++kc)
      c = wmma16(load_A(h1 + kc * 32, 256),
                 load_B(fcW2 + (size_t)cb * 256, 256, kc * 32), c);
    const int n = lane & 15, mb = (lane < 16) ? 0 : 8;
#pragma unroll
    for (int v = 0; v < 8; ++v)
      sent_emb[(size_t)(b0 + mb + v) * 256 + cb + n] = (half_t)c[v];
  }
}

// ---------------------------------------------------------------------------
// Classifier: concat(sent[256], word[128]) -> 256 -> 128 -> 2 -> log_softmax
// 16 rows / block, 8 waves.
// ---------------------------------------------------------------------------
__global__ void __launch_bounds__(256)
classifier_kernel(const half_t* __restrict__ sent_emb,     // [128][256]
                  const half_t* __restrict__ word_emb_tgt, // [128][128]
                  const half_t* __restrict__ W0, const float* __restrict__ b0_,
                  const half_t* __restrict__ W1, const float* __restrict__ b1_,
                  const float*  __restrict__ W2, const float* __restrict__ b2_,
                  float* __restrict__ out)                 // [128][2]
{
  extern __shared__ char smem[];
  half_t* xin = (half_t*)smem;          // [16][384]
  half_t* h0  = xin + 16 * 384;         // [16][256]
  half_t* h1  = h0  + 16 * 256;         // [16][128]
  float*  z   = (float*)(h1 + 16 * 128);// [16][2]

  const int tid = threadIdx.x, wave = tid >> 5, lane = tid & 31;
  const int r0 = blockIdx.x * 16;

  for (int i = tid; i < 16 * 384; i += 256) {
    const int m = i / 384, d = i % 384;
    xin[i] = (d < 256) ? sent_emb[(size_t)(r0 + m) * 256 + d]
                       : word_emb_tgt[(size_t)(r0 + m) * 128 + (d - 256)];
  }
  __syncthreads();

#pragma unroll
  for (int q = 0; q < 2; ++q) {          // layer0: N=256, K=384
    const int cb = (wave * 2 + q) * 16;
    v8f c = bias_C(b0_ + cb);
#pragma unroll
    for (int kc = 0; kc < 12; ++kc)
      c = wmma16(load_A(xin + kc * 32, 384),
                 load_B(W0 + (size_t)cb * 384, 384, kc * 32), c);
    const int n = lane & 15, mb = (lane < 16) ? 0 : 8;
#pragma unroll
    for (int v = 0; v < 8; ++v)
      h0[(mb + v) * 256 + cb + n] = (half_t)relu_f(c[v]);
  }
  __syncthreads();
  {                                       // layer1: N=128, K=256
    const int cb = wave * 16;
    v8f c = bias_C(b1_ + cb);
#pragma unroll
    for (int kc = 0; kc < 8; ++kc)
      c = wmma16(load_A(h0 + kc * 32, 256),
                 load_B(W1 + (size_t)cb * 256, 256, kc * 32), c);
    const int n = lane & 15, mb = (lane < 16) ? 0 : 8;
#pragma unroll
    for (int v = 0; v < 8; ++v)
      h1[(mb + v) * 128 + cb + n] = (half_t)relu_f(c[v]);
  }
  __syncthreads();
  if (tid < 32) {                         // 2 logits per row (VALU dot)
    const int m = tid >> 1, j = tid & 1;
    float acc = b2_[j];
    const half_t* hr = h1 + m * 128;
    const float* w = W2 + j * 128;
#pragma unroll 8
    for (int k = 0; k < 128; ++k) acc += (float)hr[k] * w[k];
    z[m * 2 + j] = acc;
  }
  __syncthreads();
  if (tid < 16) {
    const float z0 = z[tid * 2], z1 = z[tid * 2 + 1];
    const float mx = fmaxf(z0, z1);
    const float lse = mx + __logf(__expf(z0 - mx) + __expf(z1 - mx));
    out[(size_t)(r0 + tid) * 2]     = z0 - lse;
    out[(size_t)(r0 + tid) * 2 + 1] = z1 - lse;
  }
}

// ---------------------------------------------------------------------------
// Host launcher
// ---------------------------------------------------------------------------
// workspace layout (f16 element offsets)
static constexpr size_t OFF_WIH     = 0;                       // 2*2*256*128
static constexpr size_t OFF_WHH     = OFF_WIH     + 131072;    // 2*2*256*64
static constexpr size_t OFF_WFC1    = OFF_WHH     + 65536;     // 128*128
static constexpr size_t OFF_WFC2    = OFF_WFC1    + 16384;     // 128*128
static constexpr size_t OFF_SWIH0   = OFF_WFC2    + 16384;     // 2*512*128
static constexpr size_t OFF_SWIH1   = OFF_SWIH0   + 131072;    // 2*512*256
static constexpr size_t OFF_SWHH    = OFF_SWIH1   + 262144;    // 2*2*512*128
static constexpr size_t OFF_SFC1    = OFF_SWHH    + 262144;    // 256*256
static constexpr size_t OFF_SFC2    = OFF_SFC1    + 65536;     // 256*256
static constexpr size_t OFF_CW0     = OFF_SFC2    + 65536;     // 256*384
static constexpr size_t OFF_CW1     = OFF_CW0     + 98304;     // 128*256
static constexpr size_t OFF_WORDEMB = OFF_CW1     + 32768;     // 8320*128
static constexpr size_t OFF_SEQ     = OFF_WORDEMB + 1064960;   // 128*64*256
static constexpr size_t OFF_SENTEMB = OFF_SEQ     + 2097152;   // 128*256

static constexpr int SMEM_WORD = 68608 * 2 + 5120 * 4 + 64;    // 157760 B
static constexpr int SMEM_SENT = 10240 * 2 + 10240 * 4;        // 61440  B
static constexpr int SMEM_CLS  = 12288 * 2 + 32 * 4;           // 24704  B

extern "C" void kernel_launch(void* const* d_in, const int* in_sizes, int n_in,
                              void* d_out, int out_size, void* d_ws, size_t ws_size,
                              hipStream_t stream) {
  (void)in_sizes; (void)n_in; (void)out_size; (void)ws_size;
  const int*   sentences = (const int*)d_in[0];
  const int*   words     = (const int*)d_in[1];
  const float* emb       = (const float*)d_in[2];
  const float* w_Wih     = (const float*)d_in[3];
  const float* w_Whh     = (const float*)d_in[4];
  const float* w_b       = (const float*)d_in[5];
  const float* w_fc_W1   = (const float*)d_in[6];
  const float* w_fc_b1   = (const float*)d_in[7];
  const float* w_fc_W2   = (const float*)d_in[8];
  const float* w_fc_b2   = (const float*)d_in[9];
  const float* s_Wih0    = (const float*)d_in[10];
  const float* s_Wih1    = (const float*)d_in[11];
  const float* s_Whh     = (const float*)d_in[12];
  const float* s_b       = (const float*)d_in[13];
  const float* s_fc_W1   = (const float*)d_in[14];
  const float* s_fc_b1   = (const float*)d_in[15];
  const float* s_fc_W2   = (const float*)d_in[16];
  const float* s_fc_b2   = (const float*)d_in[17];
  const float* c_W0      = (const float*)d_in[18];
  const float* c_b0      = (const float*)d_in[19];
  const float* c_W1      = (const float*)d_in[20];
  const float* c_b1      = (const float*)d_in[21];
  const float* c_W2      = (const float*)d_in[22];
  const float* c_b2      = (const float*)d_in[23];
  float*  out = (float*)d_out;
  half_t* ws  = (half_t*)d_ws;

  auto cvt = [&](const float* src, size_t off, int n) {
    f32_to_f16_kernel<<<(n + 255) / 256, 256, 0, stream>>>(src, ws + off, n);
  };
  cvt(w_Wih,   OFF_WIH,   131072);
  cvt(w_Whh,   OFF_WHH,   65536);
  cvt(w_fc_W1, OFF_WFC1,  16384);
  cvt(w_fc_W2, OFF_WFC2,  16384);
  cvt(s_Wih0,  OFF_SWIH0, 131072);
  cvt(s_Wih1,  OFF_SWIH1, 262144);
  cvt(s_Whh,   OFF_SWHH,  262144);
  cvt(s_fc_W1, OFF_SFC1,  65536);
  cvt(s_fc_W2, OFF_SFC2,  65536);
  cvt(c_W0,    OFF_CW0,   98304);
  cvt(c_W1,    OFF_CW1,   32768);

  // 8320 words total (128*64 sentence slots + 128 target words), 16 per block
  word_lstm_kernel<<<520, 256, SMEM_WORD, stream>>>(
      sentences, words, emb,
      ws + OFF_WIH, ws + OFF_WHH, w_b,
      ws + OFF_WFC1, w_fc_b1, ws + OFF_WFC2, w_fc_b2,
      ws + OFF_WORDEMB);

  sent_lstm_kernel<<<8, 512, SMEM_SENT, stream>>>(
      ws + OFF_WORDEMB,
      ws + OFF_SWIH0, ws + OFF_SWIH1, ws + OFF_SWHH, s_b,
      ws + OFF_SFC1, s_fc_b1, ws + OFF_SFC2, s_fc_b2,
      ws + OFF_SEQ, ws + OFF_SENTEMB);

  classifier_kernel<<<8, 256, SMEM_CLS, stream>>>(
      ws + OFF_SENTEMB,
      ws + OFF_WORDEMB + (size_t)8192 * 128,
      ws + OFF_CW0, c_b0, ws + OFF_CW1, c_b1, c_W2, c_b2, out);
}
